// Retention_7241314861355
// MI455X (gfx1250) — matrix operand order repaired and checked by reading, not compiled, not run on hardware
//
#include <hip/hip_runtime.h>

typedef __attribute__((ext_vector_type(16))) _Float16 v16h;
typedef __attribute__((ext_vector_type(8)))  _Float16 v8h;
typedef __attribute__((ext_vector_type(8)))  float    v8f;
typedef __attribute__((ext_vector_type(4)))  float    v4f;

#define EMBED 1024
#define HEAD  1024
#define SEQL  2048
#define NBATCH 4
#define ROWS  (NBATCH * SEQL)   // 8192

__device__ __forceinline__ v16h pack2(v8h lo, v8h hi) {
    v16h r;
#pragma unroll
    for (int t = 0; t < 8; ++t) { r[t] = lo[t]; r[t + 8] = hi[t]; }
    return r;
}

// ---------------------------------------------------------------------------
// Kernel 1: P = X @ W  (f32 in, f16 out), epilogue applies XPos rotation.
// mode 0: q (scale), mode 1: k (inverse scale), mode 2: v (no rotation)
// Block = 256 threads = 8 waves (4M x 2N), block tile 128M x 64N, K-step 32.
// ---------------------------------------------------------------------------
__global__ __launch_bounds__(256) void proj_xpos(const float* __restrict__ X,
                                                 const float* __restrict__ W,
                                                 _Float16* __restrict__ Out,
                                                 int mode) {
    __shared__ _Float16 Blds[64 * 40];   // [n][k], stride 40 (pad)

    const int tid  = threadIdx.x;
    const int lane = tid & 31;
    const int wave = tid >> 5;
    const int half = lane >> 4;
    const int l16  = lane & 15;
    const int wm   = wave >> 1;          // 0..3
    const int wn   = wave & 1;           // 0..1
    const int bm0  = blockIdx.x * 128;
    const int n0   = blockIdx.y * 64;

    v8f acc[2][2];
#pragma unroll
    for (int a = 0; a < 2; ++a)
#pragma unroll
        for (int b = 0; b < 2; ++b) acc[a][b] = {};

    for (int kk = 0; kk < EMBED; kk += 32) {
        // ---- stage W tile transposed into LDS: Blds[n][k] = W[kk+k][n0+n]
        {
            const int kl = tid >> 4;          // 0..15
            const int n4 = (tid & 15) << 2;   // 0..60
#pragma unroll
            for (int p = 0; p < 2; ++p) {
                const int k = kl + 16 * p;
                v4f w = *(const v4f*)(W + (size_t)(kk + k) * HEAD + n0 + n4);
#pragma unroll
                for (int u = 0; u < 4; ++u)
                    Blds[(n4 + u) * 40 + k] = (_Float16)w[u];
            }
        }
        __syncthreads();

        // ---- A fragments straight from global (f32 -> f16 cvt)
        v16h afrag[2];
#pragma unroll
        for (int sm = 0; sm < 2; ++sm) {
            const int row = bm0 + 32 * wm + 16 * sm + l16;
            const float* ap = X + (size_t)row * EMBED + kk + 8 * half;
            v4f a0 = *(const v4f*)(ap);
            v4f a1 = *(const v4f*)(ap + 4);
            v4f a2 = *(const v4f*)(ap + 16);
            v4f a3 = *(const v4f*)(ap + 20);
            v16h a;
#pragma unroll
            for (int u = 0; u < 4; ++u) {
                a[u]      = (_Float16)a0[u];
                a[4 + u]  = (_Float16)a1[u];
                a[8 + u]  = (_Float16)a2[u];
                a[12 + u] = (_Float16)a3[u];
            }
            afrag[sm] = a;
        }
#pragma unroll
        for (int sn = 0; sn < 2; ++sn) {
            const int nl = 32 * wn + 16 * sn + l16;
            const _Float16* bp = &Blds[nl * 40 + 8 * half];
            v16h bb = pack2(*(const v8h*)bp, *(const v8h*)(bp + 16));
#pragma unroll
            for (int sm = 0; sm < 2; ++sm)
                acc[sm][sn] = __builtin_amdgcn_wmma_f32_16x16x32_f16(
                    false, afrag[sm], false, bb, (short)0, acc[sm][sn],
                    false, false);
        }
        __syncthreads();
    }

    // ---- epilogue: XPos rotary + scale, cvt f16, store
    const float LTH = 9.2103404f / 512.0f;   // ln(theta)/half
#pragma unroll
    for (int sm = 0; sm < 2; ++sm)
#pragma unroll
        for (int sn = 0; sn < 2; ++sn)
#pragma unroll
            for (int r = 0; r < 8; ++r) {
                const int row = bm0 + 32 * wm + 16 * sm + r + 8 * half;
                const int col = n0 + 32 * wn + 16 * sn + l16;
                float x  = acc[sm][sn][r];
                float xp = __shfl_xor(x, 1, 32);   // even<->odd column partner
                float y;
                if (mode == 2) {
                    y = x;
                } else {
                    const int   pos  = row & (SEQL - 1);
                    const int   ip   = col >> 1;
                    const float fpos = (float)pos;
                    const float freq = __expf(-(float)ip * LTH);
                    float sv, cv;
                    __sincosf(fpos * freq, &sv, &cv);
                    const float base = ((float)(col & ~1) + 409.6f) * (1.0f / 1433.6f);
                    float e = (fpos * (1.0f / 512.0f)) * __logf(base);
                    if (mode == 1) e = -e;   // inverse scale for k
                    const float scale = __expf(e);
                    const float cs = cv * scale, ss = sv * scale;
                    y = (col & 1) ? (x * cs + xp * ss) : (x * cs - xp * ss);
                }
                Out[(size_t)row * HEAD + col] = (_Float16)y;
            }
}

// ---------------------------------------------------------------------------
// Kernel 2: causal retention attention.
//   S = qh_i . kh_j^T  (K=1024, f16 WMMA, f32 acc), decay in registers,
//   O += S . vh_j      (K=64 via LDS round-trip)
// Grid: (L/64 i-blocks, H/256 n-blocks, B). Block 256 = 8 waves (4M x 2N).
// ---------------------------------------------------------------------------
__global__ __launch_bounds__(256) void retention_attn(
    const _Float16* __restrict__ Qh, const _Float16* __restrict__ Kh,
    const _Float16* __restrict__ Vh, float* __restrict__ Out) {
    __shared__ _Float16 Sl[64 * 72];     // S tile [i][j], stride 72
    __shared__ _Float16 Vt[256 * 72];    // vh tile transposed [n][j], stride 72

    const int tid  = threadIdx.x;
    const int lane = tid & 31;
    const int wave = tid >> 5;
    const int half = lane >> 4;
    const int l16  = lane & 15;
    const int wm   = wave >> 1;          // 0..3 -> 16 rows each
    const int wn   = wave & 1;           // 0..1 -> 128 cols each (O) / 32 cols (S)
    const int i0   = blockIdx.x * 64;
    const int nb0  = blockIdx.y * 256;
    const int rowbase = blockIdx.z * SEQL;

    const float LG = -0.03174870f;       // ln(0.96875)

    v8f oacc[8];
#pragma unroll
    for (int s = 0; s < 8; ++s) oacc[s] = {};

    for (int j0 = 0; j0 <= i0; j0 += 64) {
        if (j0 + 64 <= i0)
            __builtin_prefetch(Kh + (size_t)(rowbase + j0 + 64) * HEAD, 0, 1);

        // ---- stage vh tile transposed: Vt[n][j] = vh[j0+j][nb0+n]
        {
            const int jl = tid & 63;
            const int n8 = (tid >> 6) << 3;   // 0,8,16,24
#pragma unroll
            for (int p = 0; p < 8; ++p) {
                const int nc = p * 32 + n8;
                v8h v = *(const v8h*)(Vh + (size_t)(rowbase + j0 + jl) * HEAD + nb0 + nc);
#pragma unroll
                for (int u = 0; u < 8; ++u) Vt[(nc + u) * 72 + jl] = v[u];
            }
        }

        // ---- S = qh_i . kh_j^T, fragments direct from global (L2-resident)
        v8f sacc[2] = {{}, {}};
        for (int kk = 0; kk < HEAD; kk += 32) {
            const _Float16* ap =
                Qh + (size_t)(rowbase + i0 + 16 * wm + l16) * HEAD + kk + 8 * half;
            v16h a = pack2(*(const v8h*)ap, *(const v8h*)(ap + 16));
#pragma unroll
            for (int sn = 0; sn < 2; ++sn) {
                const _Float16* bp =
                    Kh + (size_t)(rowbase + j0 + 32 * wn + 16 * sn + l16) * HEAD +
                    kk + 8 * half;
                v16h bb = pack2(*(const v8h*)bp, *(const v8h*)(bp + 16));
                sacc[sn] = __builtin_amdgcn_wmma_f32_16x16x32_f16(
                    false, a, false, bb, (short)0, sacc[sn], false, false);
            }
        }

        // ---- decay mask + f16, into LDS
#pragma unroll
        for (int sn = 0; sn < 2; ++sn)
#pragma unroll
            for (int r = 0; r < 8; ++r) {
                const int il = 16 * wm + r + 8 * half;
                const int jl = 32 * wn + 16 * sn + l16;
                const int gi = i0 + il, gj = j0 + jl;
                const float d = (gj <= gi) ? __expf((float)(gi - gj) * LG) : 0.0f;
                Sl[il * 72 + jl] = (_Float16)(sacc[sn][r] * d);
            }
        __syncthreads();

        // ---- O += S . vh  (K = 64)
#pragma unroll
        for (int kk2 = 0; kk2 < 64; kk2 += 32) {
            const _Float16* ap = &Sl[(16 * wm + l16) * 72 + kk2 + 8 * half];
            v16h a = pack2(*(const v8h*)ap, *(const v8h*)(ap + 16));
#pragma unroll
            for (int sn2 = 0; sn2 < 8; ++sn2) {
                const _Float16* bp =
                    &Vt[(128 * wn + 16 * sn2 + l16) * 72 + kk2 + 8 * half];
                v16h bb = pack2(*(const v8h*)bp, *(const v8h*)(bp + 16));
                oacc[sn2] = __builtin_amdgcn_wmma_f32_16x16x32_f16(
                    false, a, false, bb, (short)0, oacc[sn2], false, false);
            }
        }
        __syncthreads();
    }

    // ---- write O (f32)
#pragma unroll
    for (int sn2 = 0; sn2 < 8; ++sn2)
#pragma unroll
        for (int r = 0; r < 8; ++r) {
            const int row = rowbase + i0 + 16 * wm + r + 8 * half;
            const int col = nb0 + 128 * wn + 16 * sn2 + l16;
            Out[(size_t)row * HEAD + col] = oacc[sn2][r];
        }
}

// ---------------------------------------------------------------------------
extern "C" void kernel_launch(void* const* d_in, const int* in_sizes, int n_in,
                              void* d_out, int out_size, void* d_ws, size_t ws_size,
                              hipStream_t stream) {
    const float* q  = (const float*)d_in[0];
    const float* k  = (const float*)d_in[1];
    const float* v  = (const float*)d_in[2];
    const float* Wq = (const float*)d_in[3];
    const float* Wk = (const float*)d_in[4];
    const float* Wv = (const float*)d_in[5];

    _Float16* qh = (_Float16*)d_ws;
    _Float16* kh = qh + (size_t)ROWS * HEAD;
    _Float16* vh = kh + (size_t)ROWS * HEAD;

    dim3 gp(ROWS / 128, HEAD / 64, 1), bp(256, 1, 1);
    proj_xpos<<<gp, bp, 0, stream>>>(q, Wq, qh, 0);
    proj_xpos<<<gp, bp, 0, stream>>>(k, Wk, kh, 1);
    proj_xpos<<<gp, bp, 0, stream>>>(v, Wv, vh, 2);

    dim3 ga(SEQL / 64, HEAD / 256, NBATCH), ba(256, 1, 1);
    retention_attn<<<ga, ba, 0, stream>>>(qh, kh, vh, (float*)d_out);
}
